// Attention_45930380263544
// MI455X (gfx1250) — matrix operand order repaired and checked by reading, not compiled, not run on hardware
//
#include <hip/hip_runtime.h>

typedef _Float16 v16h __attribute__((ext_vector_type(16)));
typedef _Float16 v8h  __attribute__((ext_vector_type(8)));
typedef float    v8f  __attribute__((ext_vector_type(8)));
typedef unsigned int uint32x4 __attribute__((ext_vector_type(4)));
typedef int      int32x8 __attribute__((ext_vector_type(8)));
typedef int      int32x4 __attribute__((ext_vector_type(4)));

#define BB    16
#define NN    1024
#define DIMM  384
#define HH    8
#define KDD   32
#define VDD   64
#define QKVC  1024            // H*(2*KD+VD)
#define HV    512             // H*VD
#define ATTN_SCALE 0.17677669529663687f   // 32^-0.5
#define BN_EPS 1e-5f

#if defined(__has_builtin)
#  if __has_builtin(__builtin_amdgcn_tensor_load_to_lds)
#    define HAVE_TDM 1
#  endif
#endif
#if __has_include(<hip/amd_detail/amd_gfx1250_TDM.h>)
#  define TDM_6ARG 1
#endif

// ---- workspace layout (in _Float16 elements) ------------------------------
#define OFF_Q    ((size_t)0)               //  4,194,304  q  [b*H+h][n][32]
#define OFF_K    ((size_t)4194304)         //  4,194,304  k  [b*H+h][n][32]
#define OFF_VT   ((size_t)8388608)         //  8,388,608  vT [b*H+h][vd][n]
#define OFF_HSW  ((size_t)16777216)        //  8,388,608  hardswish(out) [b*n][512]
#define OFF_X16  ((size_t)25165824)        //  6,291,456  x  f16 row-major
#define OFF_QW   ((size_t)31457280)        //    393,216  qkv_w^T f16 [1024][384]
#define OFF_PW   ((size_t)31850496)        //    196,608  proj_w^T f16 [384][512]
#define OFF_BIA  ((size_t)32047104)        //  8,388,608  bias16 [h][n][m]

// A-fragment (16x32 f16): 16 contiguous halves at p
__device__ __forceinline__ v16h ldA(const _Float16* p) {
  v8h lo = *(const v8h*)(p);
  v8h hh = *(const v8h*)(p + 8);
  v16h r;
  #pragma unroll
  for (int i = 0; i < 8; ++i) { r[i] = lo[i]; r[i + 8] = hh[i]; }
  return r;
}
// B-fragment (32x16 f16): 16 contiguous halves at p
__device__ __forceinline__ v16h ldB(const _Float16* p) {
  v8h lo = *(const v8h*)(p);
  v8h hh = *(const v8h*)(p + 8);
  v16h r;
  #pragma unroll
  for (int i = 0; i < 8; ++i) { r[i] = lo[i]; r[i + 8] = hh[i]; }
  return r;
}

// ---------------------------------------------------------------------------
// Pack kernels (bandwidth-bound, ~35MB total -> ~1.5us at 23.3 TB/s)
// ---------------------------------------------------------------------------
__global__ void cvt_x_f16(const float* __restrict__ x, _Float16* __restrict__ x16, int n) {
  int i = blockIdx.x * 256 + threadIdx.x;
  if (i < n) x16[i] = (_Float16)x[i];
}
__global__ void tr_w_f16(const float* __restrict__ w, _Float16* __restrict__ wt,
                         int K, int C) {  // w[K][C] -> wt[C][K]
  int i = blockIdx.x * 256 + threadIdx.x;
  if (i < K * C) {
    int c = i % C, kk = i / C;
    wt[(size_t)c * K + kk] = (_Float16)w[(size_t)kk * C + c];
  }
}
__global__ void gather_bias_f16(const float* __restrict__ biases, const int* __restrict__ bidx,
                                _Float16* __restrict__ b16) {
  int i = blockIdx.x * 256 + threadIdx.x;      // h*NN*NN + n*NN + m, total 8.4M
  int m = i & (NN - 1);
  int n = (i >> 10) & (NN - 1);
  int h = i >> 20;
  b16[i] = (_Float16)biases[h * NN + bidx[(size_t)n * NN + m]];
}

// ---------------------------------------------------------------------------
// Kernel A: qkv = BN(x @ qkv_w); scatter q,k (row-major) and v (transposed)
// grid (1024, 8) x 256 threads; wave -> one 16x16 tile of 16384x1024 GEMM
// ---------------------------------------------------------------------------
__global__ __launch_bounds__(256)
void qkv_gemm_bn(const _Float16* __restrict__ x16, const _Float16* __restrict__ wt,
                 const float* __restrict__ gamma, const float* __restrict__ beta,
                 const float* __restrict__ mean,  const float* __restrict__ var,
                 _Float16* __restrict__ q, _Float16* __restrict__ k,
                 _Float16* __restrict__ vT)
{
  const int lane = threadIdx.x & 31, wv = threadIdx.x >> 5;
  const int hi = lane >> 4, ln = lane & 15;
  const int rowT = blockIdx.x;              // row / 16
  const int colT = blockIdx.y * 8 + wv;     // col / 16
  const int c0 = colT * 16;
  const int row0 = rowT * 16;

  const _Float16* ar0 = x16 + (size_t)(row0 + ln) * DIMM + 16 * hi;
  const _Float16* br0 = wt  + (size_t)(c0 + ln) * DIMM + 16 * hi;

  v8f acc = {};
  for (int kb = 0; kb < DIMM; kb += 32) {
    __builtin_prefetch(br0 + kb + 64, 0, 1);
    v16h a = ldA(ar0 + kb);
    v16h b = ldB(br0 + kb);
    acc = __builtin_amdgcn_wmma_f32_16x16x32_f16(false, a, false, b,
                                                 (short)0, acc, false, false);
  }

  const int c  = c0 + ln;
  const float gs = gamma[c] * rsqrtf(var[c] + BN_EPS);
  const float mb = mean[c], bt = beta[c];
  const int head = c0 >> 7;
  const int rem  = c0 & 127;     // 16-aligned tile never straddles q/k/v segment
  const int d    = c & 127;

  #pragma unroll
  for (int vv = 0; vv < 8; ++vv) {
    const int r = row0 + vv + 8 * hi;
    const int b = r >> 10, n = r & 1023;
    const _Float16 hval = (_Float16)((acc[vv] - mb) * gs + bt);
    const size_t bh = (size_t)(b * HH + head);
    if (rem < 32)       q[(bh * NN + n) * KDD + d]             = hval;
    else if (rem < 64)  k[(bh * NN + n) * KDD + (d - 32)]      = hval;
    else                vT[(bh * VDD + (d - 64)) * NN + n]     = hval;  // transposed
  }
}

// ---------------------------------------------------------------------------
// Kernel B: per (b,h,q-tile): TDM-stage Q -> scores -> softmax -> P@V -> hsw
// grid (B*H*64) x 256 threads; LDS: 1KB qtile + 32KB f16 probs + 4KB combine
// ---------------------------------------------------------------------------
__global__ __launch_bounds__(256)
void attn_kernel(const _Float16* __restrict__ q, const _Float16* __restrict__ k,
                 const _Float16* __restrict__ vT, const _Float16* __restrict__ bias16,
                 _Float16* __restrict__ hsw)
{
  __shared__ _Float16 qtile[16 * KDD];     // 1 KB
  __shared__ _Float16 sc16[16 * NN];       // 32 KB logits -> probabilities
  __shared__ float    cmb[4 * 256];        // 4 KB split-K combine
  const int lane = threadIdx.x & 31, wv = threadIdx.x >> 5;
  const int hi = lane >> 4, ln = lane & 15;
  const int qt = blockIdx.x & 63;
  const int bh = blockIdx.x >> 6;          // b*H + h
  const int h  = bh & (HH - 1);
  const size_t base = (size_t)bh * NN;
  const _Float16* qsrc = q + (base + qt * 16) * KDD;

#if defined(HAVE_TDM)
  if (wv == 0) {
    // Tensor DMA: 16x32 f16 tile, global -> LDS (D# per ISA 8.3/8.4)
    const unsigned long long ga = (unsigned long long)(uintptr_t)qsrc;
    const unsigned lds_off = (unsigned)(uintptr_t)&qtile[0];
    uint32x4 g0;
    g0[0] = 1u;                                        // count=1, user mode
    g0[1] = lds_off;                                   // lds_addr (bytes)
    g0[2] = (unsigned)(ga & 0xffffffffu);              // global_addr[31:0]
    g0[3] = (unsigned)((ga >> 32) & 0x01ffffffu) | (2u << 30);  // addr[56:32] | type=2
    int32x8 g1;
    g1[0] = 0x10000;            // workgroup_mask=0, data_size=1 (2 bytes)
    g1[1] = (int)(32u << 16);   // tensor_dim0 = 32  (bits 79:48 low half)
    g1[2] = (int)(16u << 16);   // tensor_dim1 = 16  (bits 111:80 -> 95:80)
    g1[3] = (int)(32u << 16);   // tile_dim0 = 32    (bits 127:112)
    g1[4] = 16;                 // tile_dim1 = 16    (bits 143:128)
    g1[5] = 32;                 // tensor_dim0_stride = 32 (bits 207:160 low)
    g1[6] = 0;
    g1[7] = 0;
    int32x4 z4 = {0, 0, 0, 0};
#if defined(TDM_6ARG)
    int32x8 z8 = {0, 0, 0, 0, 0, 0, 0, 0};
    __builtin_amdgcn_tensor_load_to_lds(g0, g1, z4, z4, z8, 0);
#else
    __builtin_amdgcn_tensor_load_to_lds(g0, g1, z4, z4, 0);
#endif
    __builtin_amdgcn_s_wait_tensorcnt(0);
  }
#else
  if (wv == 0) {
    for (int i = lane; i < 16 * KDD; i += 32) qtile[i] = qsrc[i];
  }
#endif
  __syncthreads();

  // Q fragment from LDS (KD==32 == exactly one WMMA K-step)
  const v16h aq = ldA(&qtile[ln * KDD + 16 * hi]);

  // Phase 1: logits = scale * q.k^T + bias16[h]
  const _Float16* bsrc = bias16 + (size_t)h * NN * NN;
  for (int mt = wv; mt < 64; mt += 8) {
    const v16h bk = ldB(k + (base + mt * 16 + ln) * KDD + 16 * hi);
    v8f dacc = {};
    dacc = __builtin_amdgcn_wmma_f32_16x16x32_f16(false, aq, false, bk,
                                                  (short)0, dacc, false, false);
    const int key = mt * 16 + ln;
    #pragma unroll
    for (int vv = 0; vv < 8; ++vv) {
      const int r = vv + 8 * hi;
      const float bia = (float)bsrc[(size_t)(qt * 16 + r) * NN + key];
      sc16[r * NN + key] = (_Float16)(dacc[vv] * ATTN_SCALE + bia);
    }
  }
  __syncthreads();

  // Phase 2: softmax, one row per half-wave (fp32 in registers)
  {
    _Float16* row = sc16 + (2 * wv + hi) * NN;
    float m = -3.4e38f;
    for (int c = ln; c < NN; c += 16) m = fmaxf(m, (float)row[c]);
    #pragma unroll
    for (int off = 8; off > 0; off >>= 1) m = fmaxf(m, __shfl_xor(m, off, 16));
    float s = 0.f;
    for (int c = ln; c < NN; c += 16) {
      const float e = __expf((float)row[c] - m);
      row[c] = (_Float16)e;
      s += e;
    }
    #pragma unroll
    for (int off = 8; off > 0; off >>= 1) s += __shfl_xor(s, off, 16);
    const float inv = 1.f / s;
    for (int c = ln; c < NN; c += 16) row[c] = (_Float16)((float)row[c] * inv);
  }
  __syncthreads();

  // Phase 3: out = P @ V ; col tile = wv&3, K-half = wv>>2 (split-K over waves)
  const int ct = wv & 3;
  const int kh = wv >> 2;
  const _Float16* vrow = vT + ((size_t)bh * VDD + ct * 16 + ln) * NN + 16 * hi;
  v8f acc = {};
  for (int kc = kh * 16; kc < kh * 16 + 16; ++kc) {     // 16 chunks of K=32
    const v16h ap = ldA(&sc16[ln * NN + kc * 32 + 16 * hi]);
    const v16h bv = ldB(vrow + kc * 32);
    acc = __builtin_amdgcn_wmma_f32_16x16x32_f16(false, ap, false, bv,
                                                 (short)0, acc, false, false);
  }
  if (wv >= 4) {
    #pragma unroll
    for (int vv = 0; vv < 8; ++vv)
      cmb[ct * 256 + (vv + 8 * hi) * 16 + ln] = acc[vv];
  }
  __syncthreads();
  if (wv < 4) {
    const int b = bh >> 3;
    #pragma unroll
    for (int vv = 0; vv < 8; ++vv) {
      const float t = acc[vv] + cmb[ct * 256 + (vv + 8 * hi) * 16 + ln];
      const float hs = t * fminf(fmaxf(t + 3.f, 0.f), 6.f) * (1.f / 6.f);
      const int qrow = qt * 16 + vv + 8 * hi;
      hsw[((size_t)(b * NN + qrow)) * HV + h * VDD + ct * 16 + ln] = (_Float16)hs;
    }
  }
}

// ---------------------------------------------------------------------------
// Kernel C: out = BN(hardswish_out @ proj_w)  [16384x512]@[512x384] -> f32
// ---------------------------------------------------------------------------
__global__ __launch_bounds__(256)
void proj_gemm_bn(const _Float16* __restrict__ hsw, const _Float16* __restrict__ pwt,
                  const float* __restrict__ gamma, const float* __restrict__ beta,
                  const float* __restrict__ mean,  const float* __restrict__ var,
                  float* __restrict__ out)
{
  const int lane = threadIdx.x & 31, wv = threadIdx.x >> 5;
  const int hi = lane >> 4, ln = lane & 15;
  const int rowT = blockIdx.x;
  const int colT = blockIdx.y * 8 + wv;    // 0..23
  const int c0 = colT * 16;
  const int row0 = rowT * 16;

  const _Float16* ar0 = hsw + (size_t)(row0 + ln) * HV + 16 * hi;
  const _Float16* br0 = pwt + (size_t)(c0 + ln) * HV + 16 * hi;

  v8f acc = {};
  for (int kb = 0; kb < HV; kb += 32) {
    __builtin_prefetch(br0 + kb + 64, 0, 1);
    v16h a = ldA(ar0 + kb);
    v16h b = ldB(br0 + kb);
    acc = __builtin_amdgcn_wmma_f32_16x16x32_f16(false, a, false, b,
                                                 (short)0, acc, false, false);
  }

  const int c = c0 + ln;
  const float gs = gamma[c] * rsqrtf(var[c] + BN_EPS);
  const float mb = mean[c], bt = beta[c];
  #pragma unroll
  for (int vv = 0; vv < 8; ++vv) {
    const int r = row0 + vv + 8 * hi;
    out[(size_t)r * DIMM + c] = (acc[vv] - mb) * gs + bt;
  }
}

// ---------------------------------------------------------------------------
extern "C" void kernel_launch(void* const* d_in, const int* in_sizes, int n_in,
                              void* d_out, int out_size, void* d_ws, size_t ws_size,
                              hipStream_t stream)
{
  const float* x          = (const float*)d_in[0];
  const float* qkv_w      = (const float*)d_in[1];
  const float* qkv_gamma  = (const float*)d_in[2];
  const float* qkv_beta   = (const float*)d_in[3];
  const float* qkv_mean   = (const float*)d_in[4];
  const float* qkv_var    = (const float*)d_in[5];
  const float* proj_w     = (const float*)d_in[6];
  const float* proj_gamma = (const float*)d_in[7];
  const float* proj_beta  = (const float*)d_in[8];
  const float* proj_mean  = (const float*)d_in[9];
  const float* proj_var   = (const float*)d_in[10];
  const float* biases     = (const float*)d_in[11];
  const int*   bidx       = (const int*)d_in[12];

  _Float16* ws   = (_Float16*)d_ws;
  _Float16* q    = ws + OFF_Q;
  _Float16* k    = ws + OFF_K;
  _Float16* vT   = ws + OFF_VT;
  _Float16* hsw  = ws + OFF_HSW;
  _Float16* x16  = ws + OFF_X16;
  _Float16* qwt  = ws + OFF_QW;
  _Float16* pwt  = ws + OFF_PW;
  _Float16* b16  = ws + OFF_BIA;

  // pack pass
  cvt_x_f16<<<dim3((BB * NN * DIMM) / 256), 256, 0, stream>>>(x, x16, BB * NN * DIMM);
  tr_w_f16<<<dim3((DIMM * QKVC) / 256), 256, 0, stream>>>(qkv_w, qwt, DIMM, QKVC);
  tr_w_f16<<<dim3((HV * DIMM) / 256), 256, 0, stream>>>(proj_w, pwt, HV, DIMM);
  gather_bias_f16<<<dim3((HH * NN * NN) / 256), 256, 0, stream>>>(biases, bidx, b16);

  qkv_gemm_bn<<<dim3(1024, 8), 256, 0, stream>>>(
      x16, qwt, qkv_gamma, qkv_beta, qkv_mean, qkv_var, q, k, vT);

  attn_kernel<<<dim3(BB * HH * 64), 256, 0, stream>>>(q, k, vT, b16, hsw);

  proj_gemm_bn<<<dim3(1024, 3), 256, 0, stream>>>(
      hsw, pwt, proj_gamma, proj_beta, proj_mean, proj_var, (float*)d_out);
}
